// MultiHeadAttention_29798483099999
// MI455X (gfx1250) — matrix operand order repaired and checked by reading, not compiled, not run on hardware
//
#include <hip/hip_runtime.h>

// ---------------- problem constants ----------------
#define B_   4
#define S_   2048
#define D_   1024
#define H_   16
#define DH_  64
#define SP_  2080           // padded key count: 2048 keys + sink@2048 + pad -> 65*32
#define NKB_ (SP_ / 32)     // 65 key blocks of 32

typedef _Float16 v16h __attribute__((ext_vector_type(16)));
typedef _Float16 v8h  __attribute__((ext_vector_type(8)));
typedef _Float16 v4h  __attribute__((ext_vector_type(4)));
typedef float    v8f  __attribute__((ext_vector_type(8)));
typedef unsigned int v4u __attribute__((ext_vector_type(4)));
typedef int      v4i  __attribute__((ext_vector_type(4)));
typedef int      v8i  __attribute__((ext_vector_type(8)));

#if __has_builtin(__builtin_amdgcn_tensor_load_to_lds)
#define HAS_TDM 1
#else
#define HAS_TDM 0
#warning "CDNA5 probe: __builtin_amdgcn_tensor_load_to_lds NOT available - using cooperative-copy fallback"
#endif

static __device__ __forceinline__ v8h ld8(const _Float16* p) { return *(const v8h*)p; }

static __device__ __forceinline__ v16h cat16(v8h a, v8h b) {
  return __builtin_shufflevector(a, b, 0,1,2,3,4,5,6,7,8,9,10,11,12,13,14,15);
}

static __device__ __forceinline__ v8f wmma32f16(v16h a, v16h b, v8f c) {
  // D = A(16x32 f16) * B(32x16 f16) + C(16x16 f32)
  return __builtin_amdgcn_wmma_f32_16x16x32_f16(false, a, false, b, (short)0, c, false, false);
}

static __device__ __forceinline__ v8f vzero8() {
  v8f z;
#pragma unroll
  for (int i = 0; i < 8; ++i) z[i] = 0.f;
  return z;
}

#if HAS_TDM
// Issue one TDM descriptor: global -> LDS. g1 carries dims/strides (ISA 8.4).
static __device__ __forceinline__ void tdm_load(const void* gptr, unsigned lds_addr,
                                                v8i g1) {
  unsigned long long ga = (unsigned long long)(uintptr_t)gptr;
  v4u g0;
  g0[0] = 1u;                                   // count=1, is_restore=0, gather off
  g0[1] = lds_addr;                             // LDS byte address
  g0[2] = (unsigned)ga;                         // global_addr[31:0]
  g0[3] = (unsigned)((ga >> 32) & 0x01FFFFFFull) | (2u << 30);  // addr[56:32] | type=2
  v4i z4 = {0, 0, 0, 0};
#if defined(__clang_major__) && __clang_major__ >= 23
  v8i z8 = {0, 0, 0, 0, 0, 0, 0, 0};
  __builtin_amdgcn_tensor_load_to_lds(g0, g1, z4, z4, z8, 0);
#else
  __builtin_amdgcn_tensor_load_to_lds(g0, g1, z4, z4, 0);
#endif
}

// D# group1 for K block: 1-D copy of 2048 f16 (32 keys x 64 dh, contiguous)
static __device__ __forceinline__ v8i tdm_g1_k() {
  v8i g;
  g[0] = (1 << 16);            // workgroup_mask=0, data_size=1 (2 bytes)
  g[1] = (2048 << 16);         // tensor_dim0[15:0] << 16
  g[2] = (1 << 16);            // tensor_dim0 hi=0 | tensor_dim1 lo = 1
  g[3] = (2048 << 16);         // tensor_dim1 hi=0 | tile_dim0 = 2048
  g[4] = 0;                    // tile_dim1 = 0 (unused), tile_dim2 = 0
  g[5] = 2048;                 // tensor_dim0_stride lo
  g[6] = 0;
  g[7] = 0;
  return g;
}

// D# group1 for V block: 2-D tile, 64 rows (dh) x 32 f16, row stride SP_
static __device__ __forceinline__ v8i tdm_g1_v() {
  v8i g;
  g[0] = (1 << 16);            // data_size = 2 bytes
  g[1] = (SP_ << 16);          // tensor_dim0 = SP_
  g[2] = (64 << 16);           // tensor_dim1 = 64
  g[3] = (32 << 16);           // tile_dim0 = 32
  g[4] = 64;                   // tile_dim1 = 64
  g[5] = SP_;                  // tensor_dim0_stride = SP_
  g[6] = 0;
  g[7] = 0;
  return g;
}
#endif  // HAS_TDM

// ---------------- helpers: convert / fill / sink ----------------
__global__ void f32_to_f16_kernel(const float* __restrict__ in,
                                  _Float16* __restrict__ out, long n4) {
  long i  = (long)blockIdx.x * blockDim.x + threadIdx.x;
  long st = (long)gridDim.x * blockDim.x;
  for (; i < n4; i += st) {
    float4 v = ((const float4*)in)[i];
    v4h h;
    h[0] = (_Float16)v.x; h[1] = (_Float16)v.y;
    h[2] = (_Float16)v.z; h[3] = (_Float16)v.w;
    ((v4h*)out)[i] = h;
  }
}

__global__ void fill_zero_u32_kernel(unsigned int* __restrict__ p, long n) {
  long i  = (long)blockIdx.x * blockDim.x + threadIdx.x;
  long st = (long)gridDim.x * blockDim.x;
  for (; i < n; i += st) p[i] = 0u;
}

// write sink (leaky) token at key index S_ in both K and V^T buffers
__global__ void sink_init_kernel(const float* __restrict__ lk,
                                 const float* __restrict__ lv,
                                 _Float16* __restrict__ kext,
                                 _Float16* __restrict__ vT) {
  int idx = blockIdx.x * blockDim.x + threadIdx.x;
  if (idx >= B_ * D_) return;
  int b = idx / D_, n = idx % D_;
  int h = n >> 6, d = n & 63;
  kext[(((size_t)b * H_ + h) * SP_ + S_) * DH_ + d] = (_Float16)lk[n];
  vT[(((size_t)b * H_ + h) * DH_ + d) * SP_ + S_]   = (_Float16)lv[n];
}

// ---------------- GEMM: C = A[M,K](f16) * W[N,K]^T(f16) + bias[N] ----------------
// Each wave computes a 32x64 output tile (2x4 WMMA tiles), K-loop step 32.
// mode 0: out f16 row-major [M,N], scaled            (q projection, scale 1/8)
// mode 1: out f16 K-ext layout [b,h,key,64]          (k projection)
// mode 2: out f16 V^T  layout [b,h,d,SP_]            (v projection, packed stores)
// mode 3: out f32 row-major [M,N]                    (final output projection)
__global__ __launch_bounds__(256) void gemm_wmma_kernel(
    const _Float16* __restrict__ A, const _Float16* __restrict__ W,
    const float* __restrict__ bias, int M, int N, int K,
    int mode, float oscale,
    _Float16* __restrict__ oh, float* __restrict__ of) {
  const int lane = threadIdx.x & 31;
  const int hi   = lane >> 4;     // half-wave select
  const int lc   = lane & 15;

  const int tilesN = N / 64;
  const int wg = blockIdx.x * (blockDim.x >> 5) + (threadIdx.x >> 5);
  const int mt = wg / tilesN, nt = wg % tilesN;
  if (mt * 32 >= M) return;
  const int m0 = mt * 32, n0 = nt * 64;

  v8f acc[2][4];
#pragma unroll
  for (int mi = 0; mi < 2; ++mi)
#pragma unroll
    for (int ni = 0; ni < 4; ++ni) acc[mi][ni] = vzero8();

  const _Float16* arow0 = A + (size_t)(m0 + lc) * K;
  const _Float16* arow1 = A + (size_t)(m0 + 16 + lc) * K;
  const _Float16* wrow[4];
#pragma unroll
  for (int ni = 0; ni < 4; ++ni) wrow[ni] = W + (size_t)(n0 + ni * 16 + lc) * K;

  for (int kk = 0; kk < K; kk += 32) {
    const int ao = kk + (hi ? 8 : 0);    // A frag: lanes0-15 K{0..7,16..23}, lanes16-31 +8
    v16h a0 = cat16(ld8(arow0 + ao), ld8(arow0 + ao + 16));
    v16h a1 = cat16(ld8(arow1 + ao), ld8(arow1 + ao + 16));
    const int bo = kk + (hi ? 8 : 0);    // B frag: lanes0-15 K{0..7,16..23}, lanes16-31 +8
    v16h b[4];
#pragma unroll
    for (int ni = 0; ni < 4; ++ni)
      b[ni] = cat16(ld8(wrow[ni] + bo), ld8(wrow[ni] + bo + 16));
#pragma unroll
    for (int ni = 0; ni < 4; ++ni) {
      acc[0][ni] = wmma32f16(a0, b[ni], acc[0][ni]);
      acc[1][ni] = wmma32f16(a1, b[ni], acc[1][ni]);
    }
  }

  // epilogue: C layout -> lanes0-15: (M=r, N=lc), lanes16-31: (M=8+r, N=lc)
#pragma unroll
  for (int mi = 0; mi < 2; ++mi) {
#pragma unroll
    for (int ni = 0; ni < 4; ++ni) {
      const int n  = n0 + ni * 16 + lc;
      const float bn = bias[n];
      const int hh = n >> 6, dd = n & 63;
      if (mode == 2) {
        const int kb  = m0 + mi * 16 + hi * 8;   // 8 consecutive keys per lane
        const int bb  = kb / S_;
        const int key = kb % S_;
        v8h pk;
#pragma unroll
        for (int r = 0; r < 8; ++r) pk[r] = (_Float16)(acc[mi][ni][r] + bn);
        *(v8h*)(oh + ((((size_t)bb * H_ + hh) * DH_ + dd) * SP_ + key)) = pk;
      } else {
#pragma unroll
        for (int r = 0; r < 8; ++r) {
          const int m  = m0 + mi * 16 + hi * 8 + r;
          const float v = acc[mi][ni][r] + bn;
          if (mode == 0) {
            oh[(size_t)m * N + n] = (_Float16)(v * oscale);
          } else if (mode == 1) {
            const int bb = m / S_, key = m % S_;
            oh[(((size_t)bb * H_ + hh) * SP_ + key) * DH_ + dd] = (_Float16)v;
          } else {  // mode 3
            of[(size_t)m * N + n] = v;
          }
        }
      }
    }
  }
}

// ---------------- flash attention ----------------
// Block = 4 waves, one (b,h), 4 consecutive 16-query tiles. K/V blocks are staged
// into double-buffered LDS by the Tensor Data Mover (wave 0 issues, TENSORcnt
// pipelined one block ahead), shared by all 4 waves.
// S^T = K*Q^T (keys as A rows, queries as B cols); O^T = V^T * P^T.
__global__ __launch_bounds__(128) void attn_wmma_kernel(
    const _Float16* __restrict__ Q,   // [B,S,D] f16, pre-scaled by dh^-0.5
    const _Float16* __restrict__ Kx,  // [B,H,SP_,64] f16 (sink at key S_)
    const _Float16* __restrict__ Vt,  // [B,H,64,SP_] f16
    const float* __restrict__ bias,   // [B,1,1,S]
    _Float16* __restrict__ X) {       // [B,S,D] f16 attention output
  __shared__ __align__(16) _Float16 skb[2][32 * DH_];  // [key][dh]
  __shared__ __align__(16) _Float16 svb[2][DH_ * 32];  // [dh][key]

  const int lane = threadIdx.x & 31;
  const int hi   = lane >> 4;
  const int lc   = lane & 15;
  const int wave = threadIdx.x >> 5;
  const int qt   = blockIdx.x * 4 + wave;
  const int h = blockIdx.y, b = blockIdx.z;
  const int q0 = qt * 16;

  // Q as B-fragments for S^T: lane = query lc; lanes0-15 K{0..7,16..23}, lanes16-31 +8
  const _Float16* qp = Q + ((size_t)b * S_ + q0 + lc) * D_ + h * DH_;
  v16h qf[2];
#pragma unroll
  for (int t = 0; t < 2; ++t) {
    const int o = t * 32 + hi * 8;
    qf[t] = cat16(ld8(qp + o), ld8(qp + o + 16));
  }

  const _Float16* kbase = Kx + ((size_t)b * H_ + h) * (size_t)SP_ * DH_;
  const _Float16* vbase = Vt + ((size_t)b * H_ + h) * (size_t)DH_ * SP_;
  const float* bp = bias + (size_t)b * S_;

  float mrun = -1e30f, lrun = 0.f;
  v8f o_acc[4];
#pragma unroll
  for (int c = 0; c < 4; ++c) o_acc[c] = vzero8();

#if HAS_TDM
  if (wave == 0) {  // prologue: DMA block 0 into buffer 0
    tdm_load(kbase, (unsigned)(uintptr_t)(void*)&skb[0][0], tdm_g1_k());
    tdm_load(vbase, (unsigned)(uintptr_t)(void*)&svb[0][0], tdm_g1_v());
  }
#endif

  for (int kb = 0; kb < NKB_; ++kb) {
    const int k0  = kb * 32;
    const int cur = kb & 1;

#if HAS_TDM
    if (kb + 1 < NKB_) {
      if (wave == 0) {  // DMA next block into the other buffer
        const int nxt = (kb + 1) & 1;
        tdm_load(kbase + (size_t)(k0 + 32) * DH_,
                 (unsigned)(uintptr_t)(void*)&skb[nxt][0], tdm_g1_k());
        tdm_load(vbase + (size_t)(k0 + 32),
                 (unsigned)(uintptr_t)(void*)&svb[nxt][0], tdm_g1_v());
      }
      __builtin_amdgcn_s_wait_tensorcnt(2);  // oldest 2 (block kb) complete
    } else {
      __builtin_amdgcn_s_wait_tensorcnt(0);
    }
#else
    {  // fallback: cooperative copy of block kb (128 threads)
      const int t = threadIdx.x;
      const _Float16* kg = kbase + (size_t)k0 * DH_ + t * 16;
      *(v8h*)&skb[cur][t * 16 + 0] = ld8(kg);
      *(v8h*)&skb[cur][t * 16 + 8] = ld8(kg + 8);
      const int row = t >> 1, ch = (t & 1) * 16;
      const _Float16* vg = vbase + (size_t)row * SP_ + k0 + ch;
      *(v8h*)&svb[cur][row * 32 + ch + 0] = ld8(vg);
      *(v8h*)&svb[cur][row * 32 + ch + 8] = ld8(vg + 8);
    }
#endif
    __syncthreads();  // staged K/V visible to all waves

    v8f s0 = vzero8(), s1 = vzero8();
    const _Float16* kp0 = &skb[cur][lc * DH_];
    const _Float16* kp1 = &skb[cur][(16 + lc) * DH_];
#pragma unroll
    for (int t = 0; t < 2; ++t) {  // dh contraction in two K=32 steps
      const int ao = t * 32 + hi * 8;
      v16h a0 = cat16(ld8(kp0 + ao), ld8(kp0 + ao + 16));
      v16h a1 = cat16(ld8(kp1 + ao), ld8(kp1 + ao + 16));
      s0 = wmma32f16(a0, qf[t], s0);
      s1 = wmma32f16(a1, qf[t], s1);
    }

    // bias + sink (0) + pad mask, one lane per key of the block
    const int jl = k0 + lane;
    const float bl = (jl < S_) ? bp[jl] : ((jl == S_) ? 0.f : -1e30f);
#pragma unroll
    for (int r = 0; r < 8; ++r) {
      s0[r] += __shfl(bl, r + hi * 8, 32);
      s1[r] += __shfl(bl, 16 + r + hi * 8, 32);
    }

    // online softmax (per-query state lives in lane pair lc / lc+16)
    float bm = -1e30f;
#pragma unroll
    for (int r = 0; r < 8; ++r) bm = fmaxf(bm, fmaxf(s0[r], s1[r]));
    bm = fmaxf(bm, __shfl_xor(bm, 16, 32));
    const float mnew  = fmaxf(mrun, bm);
    const float scale = __expf(mrun - mnew);
    float p0[8], p1[8], ls = 0.f;
#pragma unroll
    for (int r = 0; r < 8; ++r) {
      p0[r] = __expf(s0[r] - mnew);
      p1[r] = __expf(s1[r] - mnew);
      ls += p0[r] + p1[r];
    }
    ls += __shfl_xor(ls, 16, 32);
    lrun = lrun * scale + ls;
    mrun = mnew;
#pragma unroll
    for (int c = 0; c < 4; ++c)
#pragma unroll
      for (int r = 0; r < 8; ++r) o_acc[c][r] *= scale;

    // P^T C-layout -> B-fragment via one cross-half exchange
    v16h pf;
#pragma unroll
    for (int r = 0; r < 8; ++r) {
      const float x0 = __shfl_xor(p0[r], 16, 32);
      const float x1 = __shfl_xor(p1[r], 16, 32);
      pf[r]     = (_Float16)(hi ? x1 : p0[r]);  // keys {0..7 | 16..23}
      pf[8 + r] = (_Float16)(hi ? p1[r] : x0);  // keys {8..15 | 24..31}
    }

    // O^T += V^T * P^T  (4 dh tiles)
#pragma unroll
    for (int c = 0; c < 4; ++c) {
      const _Float16* vp = &svb[cur][(c * 16 + lc) * 32];
      const int ao = hi * 16;
      v16h a = cat16(ld8(vp + ao), ld8(vp + ao + 8));
      o_acc[c] = wmma32f16(a, pf, o_acc[c]);
    }

    __syncthreads();  // all waves done with buf[cur] before TDM overwrites it
  }

  const float inv = 1.f / lrun;
  _Float16* xp = X + ((size_t)b * S_ + q0 + lc) * D_ + h * DH_;
#pragma unroll
  for (int c = 0; c < 4; ++c)
#pragma unroll
    for (int r = 0; r < 8; ++r)
      xp[c * 16 + hi * 8 + r] = (_Float16)(o_acc[c][r] * inv);
}

// ---------------- host: launch sequence ----------------
extern "C" void kernel_launch(void* const* d_in, const int* in_sizes, int n_in,
                              void* d_out, int out_size, void* d_ws, size_t ws_size,
                              hipStream_t stream) {
  (void)in_sizes; (void)n_in; (void)out_size; (void)ws_size;
  const float* query = (const float*)d_in[0];
  const float* bias  = (const float*)d_in[1];
  const float* Wq = (const float*)d_in[2];
  const float* bq = (const float*)d_in[3];
  const float* Wk = (const float*)d_in[4];
  const float* bk = (const float*)d_in[5];
  const float* Wv = (const float*)d_in[6];
  const float* bv = (const float*)d_in[7];
  const float* Wo = (const float*)d_in[8];
  const float* bo = (const float*)d_in[9];
  const float* lk = (const float*)d_in[10];
  const float* lv = (const float*)d_in[11];
  float* out = (float*)d_out;

  // workspace partition (halves); x_f16 aliases act (act dead after QKV GEMMs)
  _Float16* ws = (_Float16*)d_ws;
  const size_t nAct = (size_t)B_ * S_ * D_;         // 8,388,608
  const size_t nW   = (size_t)D_ * D_;              // 1,048,576
  const size_t nKV  = (size_t)B_ * H_ * SP_ * DH_;  // 8,519,680
  size_t off = 0;
  _Float16* act = ws + off; off += nAct;
  _Float16* wqh = ws + off; off += nW;
  _Float16* wkh = ws + off; off += nW;
  _Float16* wvh = ws + off; off += nW;
  _Float16* woh = ws + off; off += nW;
  _Float16* qh  = ws + off; off += nAct;
  _Float16* kxt = ws + off; off += nKV;
  _Float16* vtt = ws + off; off += nKV;
  _Float16* xh  = act;  // alias

  auto cvt = [&](const float* in, _Float16* o, long n) {
    long n4 = n / 4;
    int blocks = (int)((n4 + 255) / 256);
    if (blocks > 4096) blocks = 4096;
    f32_to_f16_kernel<<<blocks, 256, 0, stream>>>(in, o, n4);
  };
  cvt(query, act, (long)nAct);
  cvt(Wq, wqh, (long)nW);
  cvt(Wk, wkh, (long)nW);
  cvt(Wv, wvh, (long)nW);
  cvt(Wo, woh, (long)nW);

  {  // zero K-ext and V^T (pads must be clean: 0xAA f16 could be NaN -> 0*NaN=NaN)
    long n32 = (long)(nKV / 2);
    fill_zero_u32_kernel<<<4096, 256, 0, stream>>>((unsigned int*)kxt, n32);
    fill_zero_u32_kernel<<<4096, 256, 0, stream>>>((unsigned int*)vtt, n32);
  }
  sink_init_kernel<<<(B_ * D_ + 255) / 256, 256, 0, stream>>>(lk, lv, kxt, vtt);

  const int M = B_ * S_, N = D_, K = D_;
  const int waves  = (M / 32) * (N / 64);
  const int blocks = waves / 8;  // 256 threads = 8 waves per block

  // q = (query Wq^T + bq) * dh^-0.5  -> [M,N] f16
  gemm_wmma_kernel<<<blocks, 256, 0, stream>>>(act, wqh, bq, M, N, K, 0, 0.125f, qh, nullptr);
  // k -> K-ext layout [b,h,key,64]
  gemm_wmma_kernel<<<blocks, 256, 0, stream>>>(act, wkh, bk, M, N, K, 1, 1.0f, kxt, nullptr);
  // v -> V^T layout [b,h,d,SP_]
  gemm_wmma_kernel<<<blocks, 256, 0, stream>>>(act, wvh, bv, M, N, K, 2, 1.0f, vtt, nullptr);

  // attention: grid (qtiles/4, H, B), 128 threads = 4 waves, wave = one 16-query tile
  attn_wmma_kernel<<<dim3(S_ / 16 / 4, H_, B_), 128, 0, stream>>>(qh, kxt, vtt, bias, xh);

  // out = x Wo^T + bo (f32)
  gemm_wmma_kernel<<<blocks, 256, 0, stream>>>(xh, woh, bo, M, N, K, 3, 1.0f, nullptr, out);
}
